// DeformableConv2d_2551210574705
// MI455X (gfx1250) — matrix operand order repaired and checked by reading, not compiled, not run on hardware
//
#include <hip/hip_runtime.h>

typedef __attribute__((ext_vector_type(16))) _Float16 v16h;
typedef __attribute__((ext_vector_type(8)))  _Float16 v8h;
typedef __attribute__((ext_vector_type(8)))  float    v8f;

#define CIN    96
#define COUT   96
#define HH     128
#define WW     128
#define TILE_W 128
#define NPOS   128
#define SPAD   104   // padded cin stride (halves) for sample matrix: 208B rows, 16B aligned
#define XTW    132   // padded halo width (130 used)

// One workgroup: one full output row (b, h, w=0..127), 512 threads = 16 wave32.
__global__ __launch_bounds__(512, 1)
void dcn_kernel(const float* __restrict__ x,        const float* __restrict__ weight,
                const float* __restrict__ bias,     const float* __restrict__ offset_w,
                const float* __restrict__ offset_b, const float* __restrict__ mask_w,
                const float* __restrict__ mask_b,   float* __restrict__ out)
{
    // 6 co-blocks x 9 taps x 3 K-chunks of pre-swizzled 16x32 f16 A fragments
    __shared__ __align__(32) _Float16 Wfrag[6 * 9 * 3 * 32 * 16];   // 165888 B
    __shared__ _Float16 xt[CIN * 3 * XTW];                          //  76032 B (f16 halo for offset conv)
    __shared__ float red[NPOS * 28];                                //  14336 B (atomic conv reduction)
    __shared__ float offY[9 * NPOS], offX[9 * NPOS], mval[9 * NPOS]; // 13824 B
    __shared__ __align__(32) _Float16 S[NPOS * SPAD];               //  26624 B (sampled B matrix, f16)

    const int t    = threadIdx.x;
    const int lane = t & 31;
    const int wid  = t >> 5;
    const int b    = blockIdx.z;
    const int h    = blockIdx.y;

    // ---- Phase 0a: build f16 A-fragments of main weight in WMMA lane layout ----
    // A 16x32 f16 layout: lane L holds M = L&15; half hh -> K = (hh&7) + (hh<8?0:16) + (L<16?0:8)
    for (int idx = t; idx < 6 * 9 * 3 * 32 * 16; idx += 512) {
        int hh  = idx & 15;
        int ln  = (idx >> 4) & 31;
        int rem = idx >> 9;              // (cb*9 + k)*3 + c
        int c   = rem % 3;
        int kk  = (rem / 3) % 9;
        int cb  = rem / 27;
        int K   = (hh & 7) + ((hh >> 3) << 4) + ((ln >> 4) << 3);
        int cin = c * 32 + K;
        int co  = cb * 16 + (ln & 15);
        int ky  = kk / 3, kx = kk % 3;
        Wfrag[idx] = (_Float16)weight[((co * CIN + cin) * 3 + ky) * 3 + kx];
    }

    // ---- Phase 0b: stage x halo (96 x 3 x 130, zero-padded) as f16 ----
    for (int i = t; i < CIN * 3 * XTW; i += 512) {
        int c  = i % XTW;
        int rr = (i / XTW) % 3;
        int ci = i / (3 * XTW);
        float v = 0.f;
        int gx = c - 1, gy = h - 1 + rr;
        if (c < 130 && gx >= 0 && gx < WW && gy >= 0 && gy < HH)
            v = x[(((size_t)b * CIN + ci) * HH + gy) * WW + gx];
        xt[i] = (_Float16)v;
    }
    // ---- Phase 0c: zero the reduction buffer ----
    for (int i = t; i < NPOS * 28; i += 512) red[i] = 0.f;
    __syncthreads();

    // ---- Phase A: 3x3 conv -> 18 offsets + 9 mask logits per pixel ----
    const int pos  = t & (NPOS - 1);
    const int cin0 = __builtin_amdgcn_readfirstlane(t >> 7) * 24;  // wave-uniform Cin quarter
    float acc[27];
#pragma unroll
    for (int o = 0; o < 27; ++o) acc[o] = 0.f;
    for (int ci = 0; ci < 24; ++ci) {
        const int cc = cin0 + ci;
#pragma unroll
        for (int r = 0; r < 3; ++r) {
            const _Float16* xr = &xt[(cc * 3 + r) * XTW + pos];
            float xa = (float)xr[0], xm = (float)xr[1], xc = (float)xr[2];
            const float* wo = offset_w + (cc * 3 + r) * 3;  // wave-uniform addr -> scalar loads
            const float* wm = mask_w   + (cc * 3 + r) * 3;
#pragma unroll
            for (int o = 0; o < 18; ++o)
                acc[o] += xa * wo[o * 864] + xm * wo[o * 864 + 1] + xc * wo[o * 864 + 2];
#pragma unroll
            for (int o = 0; o < 9; ++o)
                acc[18 + o] += xa * wm[o * 864] + xm * wm[o * 864 + 1] + xc * wm[o * 864 + 2];
        }
    }
#pragma unroll
    for (int o = 0; o < 27; ++o)
        atomicAdd(&red[pos * 28 + o], acc[o]);   // ds_add_f32, 4 partials per pixel
    __syncthreads();
    if (t < NPOS) {
#pragma unroll
        for (int k = 0; k < 9; ++k) {
            offY[k * NPOS + t] = red[t * 28 + 2 * k]     + offset_b[2 * k];
            offX[k * NPOS + t] = red[t * 28 + 2 * k + 1] + offset_b[2 * k + 1];
            float z = red[t * 28 + 18 + k] + mask_b[k];
            mval[k * NPOS + t] = 1.f / (1.f + __expf(-z));
        }
    }
    __syncthreads();

    // ---- Phase B: per tap, bilinear sample -> LDS f16, then WMMA accumulate ----
    v8f facc[3] = {};
    const float* xb = x + ((size_t)b * CIN + cin0) * (HH * WW);
    for (int k = 0; k < 9; ++k) {
        float dy = offY[k * NPOS + pos], dx = offX[k * NPOS + pos], m = mval[k * NPOS + pos];
        float py = (float)(h - 1 + k / 3) + dy;
        float px = (float)(pos - 1 + k % 3) + dx;
        float y0f = floorf(py), x0f = floorf(px);
        float wy = py - y0f, wx = px - x0f;
        int y0 = (int)y0f, x0i = (int)x0f;
        int y1 = y0 + 1, x1i = x0i + 1;
        float vy0 = (y0  >= 0 && y0  < HH) ? 1.f : 0.f;
        float vy1 = (y1  >= 0 && y1  < HH) ? 1.f : 0.f;
        float vx0 = (x0i >= 0 && x0i < WW) ? 1.f : 0.f;
        float vx1 = (x1i >= 0 && x1i < WW) ? 1.f : 0.f;
        float w00 = (1.f - wy) * (1.f - wx) * m * vy0 * vx0;
        float w01 = (1.f - wy) * wx         * m * vy0 * vx1;
        float w10 = wy         * (1.f - wx) * m * vy1 * vx0;
        float w11 = wy         * wx         * m * vy1 * vx1;
        int y0c = min(max(y0, 0), HH - 1), y1c = min(max(y1, 0), HH - 1);
        int x0c = min(max(x0i, 0), WW - 1), x1c = min(max(x1i, 0), WW - 1);
        int i00 = y0c * WW + x0c, i01 = y0c * WW + x1c;
        int i10 = y1c * WW + x0c, i11 = y1c * WW + x1c;
#pragma unroll 8
        for (int ci = 0; ci < 24; ++ci) {
            const float* p = xb + ci * (HH * WW);
            float v = w00 * p[i00] + w01 * p[i01] + w10 * p[i10] + w11 * p[i11];
            S[pos * SPAD + cin0 + ci] = (_Float16)v;
        }

        // Prefetch next tap's gather corners into near caches (global_prefetch_b8);
        // overlaps with the WMMA phase below and next tap's address setup.
        if (k < 8) {
            float dy2 = offY[(k + 1) * NPOS + pos], dx2 = offX[(k + 1) * NPOS + pos];
            float py2 = (float)(h - 1 + (k + 1) / 3) + dy2;
            float px2 = (float)(pos - 1 + (k + 1) % 3) + dx2;
            int y0n = (int)floorf(py2), x0n = (int)floorf(px2);
            int y0cn = min(max(y0n, 0), HH - 1), y1cn = min(max(y0n + 1, 0), HH - 1);
            int x0cn = min(max(x0n, 0), WW - 1);
            int i00n = y0cn * WW + x0cn, i10n = y1cn * WW + x0cn;
#pragma unroll 8
            for (int ci = 0; ci < 24; ++ci) {
                const float* p = xb + ci * (HH * WW);
                __builtin_prefetch(p + i00n, 0, 3);
                __builtin_prefetch(p + i10n, 0, 3);
            }
        }
        __syncthreads();

        // Each wave: 3 of 48 (co_blk, ntile) tiles x 3 K-chunks of 16x16x32 f16 WMMA
#pragma unroll
        for (int j = 0; j < 3; ++j) {
            int tau = wid + j * 16;
            int cb  = tau % 6, nt = tau / 6;
#pragma unroll
            for (int c = 0; c < 3; ++c) {
                const v16h a = *(const v16h*)&Wfrag[(((cb * 9 + k) * 3 + c) * 32 + lane) * 16];
                int sbase = (nt * 16 + (lane & 15)) * SPAD + c * 32 + ((lane >> 4) << 3);
                v8h blo = *(const v8h*)&S[sbase];
                v8h bhi = *(const v8h*)&S[sbase + 16];
                v16h bb = __builtin_shufflevector(blo, bhi,
                    0, 1, 2, 3, 4, 5, 6, 7, 8, 9, 10, 11, 12, 13, 14, 15);
                facc[j] = __builtin_amdgcn_wmma_f32_16x16x32_f16(
                    false, a, false, bb, (short)0, facc[j], false, false);
            }
        }
        __syncthreads();  // protect S before next tap overwrites it
    }

    // ---- Epilogue: scatter D tiles (+bias). C/D layout: VGPR r -> M = r + (lane>=16 ? 8:0), N = lane&15
#pragma unroll
    for (int j = 0; j < 3; ++j) {
        int tau  = wid + j * 16;
        int cb   = tau % 6, nt = tau / 6;
        int wcol = nt * 16 + (lane & 15);
        int mo   = (lane >= 16) ? 8 : 0;
#pragma unroll
        for (int r = 0; r < 8; ++r) {
            int co = cb * 16 + r + mo;
            out[(((size_t)b * COUT + co) * HH + h) * WW + wcol] = facc[j][r] + bias[co];
        }
    }
}

extern "C" void kernel_launch(void* const* d_in, const int* in_sizes, int n_in,
                              void* d_out, int out_size, void* d_ws, size_t ws_size,
                              hipStream_t stream) {
    (void)in_sizes; (void)n_in; (void)d_ws; (void)ws_size; (void)out_size;
    const float* x        = (const float*)d_in[0];
    const float* weight   = (const float*)d_in[1];
    const float* bias     = (const float*)d_in[2];
    const float* offset_w = (const float*)d_in[3];
    const float* offset_b = (const float*)d_in[4];
    const float* mask_w   = (const float*)d_in[5];
    const float* mask_b   = (const float*)d_in[6];
    float* out = (float*)d_out;
    dim3 grid(1, HH, 16);  // 2048 workgroups, one per (batch, row)
    dcn_kernel<<<grid, 512, 0, stream>>>(x, weight, bias, offset_w, offset_b,
                                         mask_w, mask_b, out);
}
